// DeformableConvAttentionClsLoc_41824391529105
// MI455X (gfx1250) — matrix operand order repaired and checked by reading, not compiled, not run on hardware
//
#include <hip/hip_runtime.h>
#include <math.h>

// ---------------------------------------------------------------------------
// DeformableConvAttentionClsLoc for MI455X (gfx1250, wave32, WMMA).
// All heavy GEMMs use v_wmma_f32_16x16x32_bf16 (f32 accumulate, bf16 A/B).
// LDS tiles are stored pre-swizzled into the ISA 7.12.2 fragment layout so
// each lane loads its 16-element fragment with two ds_load_b128.
// NOTE: the GEMM assumes M%128==0, N%128==0, K%32==0 — true for every call
// in this pipeline (no bounds guards -> no exec-mask churn in the staging).
// ---------------------------------------------------------------------------

#define B_   8
#define NH_  8
#define HC_  32
#define NG_  4
#define NC_  256
#define CG_  64
#define H_   64
#define W_   64
#define HW_  4096
#define KM_  9
#define NS_  2048
#define DIMOUT_ 256
#define SCALE_ 0.17677669529663687f  // 32^-0.5
#define ORF_H (2.0f / 64.0f)         // ORF/H = ORF/W

typedef __attribute__((ext_vector_type(16))) __bf16 v16bf;
typedef __attribute__((ext_vector_type(8)))  __bf16 v8bf;
typedef __attribute__((ext_vector_type(2)))  __bf16 v2bf;
typedef __attribute__((ext_vector_type(8)))  float  v8f;

__device__ __forceinline__ float gelu_exact(float x) {
  return 0.5f * x * (1.0f + erff(x * 0.70710678118654752440f));
}

// ---------------------------------------------------------------------------
// Generic WMMA GEMM:  Y[m,n] = act( sum_k A[m,k]*B[k,n] + bias )
//   A: f32, A[m,k] = A[ab + m*lda + k]            (k-contiguous)
//   B: f32, B[k,n] = B[bb + k*ldbK + n*ldbN]
//   Y: f32 or bf16,  Y[yb + m*ldy + n]
//   biasMode: 0=none 1=bias[m] 2=bias[n];  act: 0=none 1=gelu 2=relu
// Block = 256 threads (8 waves). Tile 128(M) x 128(N) x 32(K).
//
// LDS holds fragments directly: sXf[tile][half][lane][8] __bf16, so a lane's
// v16bf operand is two contiguous 16B LDS reads.
//   A (16x32 bf16): lane = (m&15) + 16*((k>>3)&1); elem pair base 2*j where
//                   j = k'<8 ? k'/2 : 4+(k'-16)/2, k' = k - 8*((k>>3)&1)
//   B (32x16 bf16): lane = (n&15) + 16*((k>>3)&1); elem = (k&7) + 8*(k>>4)
// ---------------------------------------------------------------------------
__global__ __launch_bounds__(256)
void gemm_bf16_wmma(const float* __restrict__ A, long long aBatch, int lda,
                    const float* __restrict__ Bm, long long bBatch, int ldbK, int ldbN,
                    void* __restrict__ Y, long long yBatch, int ldy,
                    const float* __restrict__ bias, int biasMode, int act, int outBf16,
                    int M, int N, int Kd)
{
  __shared__ __align__(16) __bf16 sAf[8][2][32][8];   // 8 KB
  __shared__ __align__(16) __bf16 sBf[8][2][32][8];   // 8 KB

  const int tid  = threadIdx.x;
  const int wave = tid >> 5;
  const int lane = tid & 31;
  const int lo16 = lane & 15;
  const int hi   = lane >> 4;
  const int M0 = blockIdx.y * 128;
  const int N0 = blockIdx.x * 128;
  const long long ab = (long long)blockIdx.z * aBatch;
  const long long bb = (long long)blockIdx.z * bBatch;
  const long long yb = (long long)blockIdx.z * yBatch;

  // Precompute staging coordinates (loop-invariant).
  // A side: pair id = i*256+tid -> row r, even k "kp"
  // B side: pair id = i*256+tid -> col n, even k "k"
  v8f acc[8];
  for (int i = 0; i < 8; ++i)
    for (int j = 0; j < 8; ++j) acc[i][j] = 0.0f;

  for (int kt = 0; kt < Kd; kt += 32) {
    // ---- A tile (128 x 32): 2048 even-k float2 loads, fragment layout
    #pragma unroll
    for (int i = 0; i < 8; ++i) {
      int e = i * 256 + tid;            // pair id: 128 rows x 16 k-pairs
      int r  = e >> 4;
      int kp = (e & 15) * 2;            // even k within tile
      float2 v = *(const float2*)(A + ab + (long long)(M0 + r) * lda + kt + kp);
      int hi2 = (kp >> 3) & 1;
      int k2  = kp - hi2 * 8;                       // {0,2,4,6,16,18,20,22}
      int j   = (k2 < 8) ? (k2 >> 1) : (4 + ((k2 - 16) >> 1));
      int ii  = 2 * j;                              // even elem index 0..14
      v2bf pk; pk[0] = (__bf16)v.x; pk[1] = (__bf16)v.y;
      *(v2bf*)&sAf[r >> 4][ii >> 3][(r & 15) + 16 * hi2][ii & 7] = pk;
    }
    // ---- B tile (32 x 128): 2048 even-k pairs, fragment layout
    #pragma unroll
    for (int i = 0; i < 8; ++i) {
      int e = i * 256 + tid;            // pair id: 16 k-pairs x 128 n
      int n  = e & 127;
      int k  = (e >> 7) * 2;            // even k within tile
      long long base = bb + (long long)(kt + k) * ldbK + (long long)(N0 + n) * ldbN;
      float v0 = Bm[base];
      float v1 = Bm[base + ldbK];
      int i0 = (k & 7) + 8 * (k >> 4);              // even elem index 0..14
      v2bf pk; pk[0] = (__bf16)v0; pk[1] = (__bf16)v1;
      *(v2bf*)&sBf[n >> 4][i0 >> 3][(n & 15) + 16 * ((k >> 3) & 1)][i0 & 7] = pk;
    }
    __syncthreads();

    // ---- fragments: two b128 LDS reads each; load all operands, then
    //      issue the 8 WMMAs back-to-back (single dscnt wait).
    v8bf a0 = *(const v8bf*)&sAf[wave][0][lane][0];
    v8bf a1 = *(const v8bf*)&sAf[wave][1][lane][0];
    v16bf afrag = __builtin_shufflevector(a0, a1,
        0,1,2,3,4,5,6,7,8,9,10,11,12,13,14,15);

    v16bf bfr[8];
    #pragma unroll
    for (int ns = 0; ns < 8; ++ns) {
      v8bf b0 = *(const v8bf*)&sBf[ns][0][lane][0];
      v8bf b1 = *(const v8bf*)&sBf[ns][1][lane][0];
      bfr[ns] = __builtin_shufflevector(b0, b1,
          0,1,2,3,4,5,6,7,8,9,10,11,12,13,14,15);
    }
    #pragma unroll
    for (int ns = 0; ns < 8; ++ns) {
      acc[ns] = __builtin_amdgcn_wmma_f32_16x16x32_bf16(
          false, afrag, false, bfr[ns], (short)0, acc[ns], false, false);
    }
    __syncthreads();
  }

  // Epilogue. C/D layout: lane row = hi*8 + vgpr, col = lo16.
  #pragma unroll
  for (int ns = 0; ns < 8; ++ns) {
    #pragma unroll
    for (int r = 0; r < 8; ++r) {
      int m = M0 + wave * 16 + hi * 8 + r;
      int n = N0 + ns * 16 + lo16;
      float v = acc[ns][r];
      if (biasMode == 1) v += bias[m];
      else if (biasMode == 2) v += bias[n];
      if (act == 1) v = gelu_exact(v);
      else if (act == 2) v = fmaxf(v, 0.0f);
      long long o = yb + (long long)m * ldy + n;
      if (outBf16) ((__bf16*)Y)[o] = (__bf16)v;
      else         ((float*)Y)[o]  = v;
    }
  }
}

// ---------------------------------------------------------------------------
// Depthwise 3x3 conv ('SAME', zero pad) + gelu over qf viewed as (32,64,H,W)
// ---------------------------------------------------------------------------
__global__ void dwconv_kernel(const float* __restrict__ qf,
                              const float* __restrict__ Wdw,
                              const float* __restrict__ bdw,
                              float* __restrict__ hbuf)
{
  int t = blockIdx.x * 256 + threadIdx.x;        // 32*64*4096
  int x  = t & 63;
  int y  = (t >> 6) & 63;
  int c  = (t >> 12) & 63;
  int bg = t >> 18;
  int b = bg >> 2, g = bg & 3;
  const float* src = qf + ((long long)(b * NC_ + g * CG_ + c)) * HW_;
  float acc = bdw[c];
  #pragma unroll
  for (int ky = 0; ky < 3; ++ky) {
    int yy = y + ky - 1;
    if (yy < 0 || yy > 63) continue;
    #pragma unroll
    for (int kx = 0; kx < 3; ++kx) {
      int xx = x + kx - 1;
      if (xx < 0 || xx > 63) continue;
      acc += src[yy * 64 + xx] * Wdw[c * 9 + ky * 3 + kx];
    }
  }
  hbuf[t] = gelu_exact(acc);
}

// ---------------------------------------------------------------------------
// Offset net: 18-ch pointwise conv + tanh*ORF/H + reference grid -> pos
// pos layout (bg, h, w, km, 2); element0=y, element1=x
// ---------------------------------------------------------------------------
__global__ void offset_kernel(const float* __restrict__ hbuf,
                              const float* __restrict__ Wpw,
                              float* __restrict__ pos)
{
  __shared__ float wpw[18 * 64];
  for (int i = threadIdx.x; i < 18 * 64; i += 256) wpw[i] = Wpw[i];
  __syncthreads();
  int t = blockIdx.x * 256 + threadIdx.x;        // 32*4096
  int hw = t & 4095;
  int bg = t >> 12;
  int y = hw >> 6, x = hw & 63;
  const float* src = hbuf + (long long)bg * CG_ * HW_ + hw;
  float acc[18];
  #pragma unroll
  for (int o = 0; o < 18; ++o) acc[o] = 0.0f;
  for (int c = 0; c < 64; ++c) {
    float hv = src[(long long)c * HW_];
    #pragma unroll
    for (int o = 0; o < 18; ++o) acc[o] += hv * wpw[o * 64 + c];
  }
  float* dst = pos + ((long long)t) * 18;
  #pragma unroll
  for (int k = 0; k < 9; ++k) {
    int ky = k / 3, kx = k % 3;
    int cy = min(max(y + ky - 1, 0), 63);
    int cx = min(max(x + kx - 1, 0), 63);
    float ry = (cy + 0.5f) * 0.03125f - 1.0f;
    float rx = (cx + 0.5f) * 0.03125f - 1.0f;
    dst[k * 2 + 0] = tanhf(acc[2 * k + 0]) * ORF_H + ry;
    dst[k * 2 + 1] = tanhf(acc[2 * k + 1]) * ORF_H + rx;
  }
}

// ---------------------------------------------------------------------------
// q gather: q[b,c,n] = qf[b,c,idx_y[n],idx_x[n]]
// ---------------------------------------------------------------------------
__global__ void qgather_kernel(const float* __restrict__ qf,
                               const int* __restrict__ idx,
                               float* __restrict__ q)
{
  int t = blockIdx.x * 256 + threadIdx.x;        // 8*256*2048
  int n = t & 2047;
  int bc = t >> 11;
  int iy = idx[n * 2 + 0], ix = idx[n * 2 + 1];
  q[t] = qf[(long long)bc * HW_ + iy * 64 + ix];
}

// ---------------------------------------------------------------------------
// Bilinear sampling + the reference's reshape scramble:
//   p = km'*4096 + iy*64 + ix ;  h=p/576, w=(p%576)/9, km=p%9
//   x_sampled[b, g*64+cg, km', n] = bilerp(x1[b, g*64+cg], pos[bg,h,w,km])
// ---------------------------------------------------------------------------
__global__ void sample_kernel(const float* __restrict__ x1,
                              const float* __restrict__ pos,
                              const int* __restrict__ idx,
                              float* __restrict__ xs)     // d_out + 2
{
  int t = blockIdx.x * 256 + threadIdx.x;        // 8*4*9*2048
  int n = t & 2047;
  int rest = t >> 11;
  int kmp = rest % 9;
  int rest2 = rest / 9;
  int g = rest2 & 3;
  int b = rest2 >> 2;
  int bg = b * 4 + g;

  int iy = idx[n * 2 + 0], ix = idx[n * 2 + 1];
  int p = kmp * 4096 + iy * 64 + ix;
  int h = p / 576;
  int r = p - h * 576;
  int w = r / 9;
  int km = r - w * 9;

  const float* pp = pos + (((long long)bg * 4096 + h * 64 + w) * 9 + km) * 2;
  float gy = pp[0], gx = pp[1];               // grid[...,::-1]: x from [1]
  float ixf = (gx + 1.0f) * 0.5f * 63.0f;
  float iyf = (gy + 1.0f) * 0.5f * 63.0f;
  float x0f = floorf(ixf), y0f = floorf(iyf);
  float wx = ixf - x0f, wy = iyf - y0f;

  float vx0 = (x0f >= 0.0f && x0f <= 63.0f) ? 1.0f : 0.0f;
  float vx1 = (x0f + 1.0f >= 0.0f && x0f + 1.0f <= 63.0f) ? 1.0f : 0.0f;
  float vy0 = (y0f >= 0.0f && y0f <= 63.0f) ? 1.0f : 0.0f;
  float vy1 = (y0f + 1.0f >= 0.0f && y0f + 1.0f <= 63.0f) ? 1.0f : 0.0f;
  int xc0 = (int)fminf(fmaxf(x0f, 0.0f), 63.0f);
  int xc1 = (int)fminf(fmaxf(x0f + 1.0f, 0.0f), 63.0f);
  int yc0 = (int)fminf(fmaxf(y0f, 0.0f), 63.0f);
  int yc1 = (int)fminf(fmaxf(y0f + 1.0f, 0.0f), 63.0f);

  float w00 = (1.0f - wx) * (1.0f - wy) * vx0 * vy0;
  float w01 = wx * (1.0f - wy) * vx1 * vy0;
  float w10 = (1.0f - wx) * wy * vx0 * vy1;
  float w11 = wx * wy * vx1 * vy1;

  const float* img = x1 + (long long)(b * NC_ + g * CG_) * HW_;
  int i00 = yc0 * 64 + xc0, i01 = yc0 * 64 + xc1;
  int i10 = yc1 * 64 + xc0, i11 = yc1 * 64 + xc1;
  long long obase = (((long long)(b * NC_ + g * CG_) * 9 + kmp) * 2048) + n;
  for (int cg = 0; cg < 64; ++cg) {
    const float* ch = img + (long long)cg * HW_;
    float v = w00 * ch[i00] + w01 * ch[i01] + w10 * ch[i10] + w11 * ch[i11];
    xs[obase + (long long)cg * 9 * 2048] = v;
  }
}

// ---------------------------------------------------------------------------
// Attention: 9-tap softmax per (head, position); writes out_mat and attn mean
// ---------------------------------------------------------------------------
__global__ void attention_kernel(const float* __restrict__ q,
                                 const __bf16* __restrict__ kk,
                                 const __bf16* __restrict__ vv,
                                 float* __restrict__ out_mat,
                                 float* __restrict__ attn_mean)
{
  int t = blockIdx.x * 256 + threadIdx.x;        // 64*2048
  int n = t & 2047;
  int bh = t >> 11;
  int b = bh >> 3, nh = bh & 7;
  int cbase = b * NC_ + nh * HC_;

  float qv[HC_];
  #pragma unroll
  for (int h = 0; h < HC_; ++h) qv[h] = q[(long long)(cbase + h) * 2048 + n];

  float lo[9];
  float mx = -1e30f;
  #pragma unroll
  for (int km = 0; km < 9; ++km) {
    float s = 0.0f;
    for (int h = 0; h < HC_; ++h) {
      long long ki = ((long long)(cbase + h) * 9 + km) * 2048 + n;
      s += qv[h] * (float)kk[ki];
    }
    s *= SCALE_;
    lo[km] = s;
    mx = fmaxf(mx, s);
  }
  float se = 0.0f;
  #pragma unroll
  for (int km = 0; km < 9; ++km) { lo[km] = expf(lo[km] - mx); se += lo[km]; }
  float inv = 1.0f / se;
  #pragma unroll
  for (int km = 0; km < 9; ++km) {
    lo[km] *= inv;
    atomicAdd(&attn_mean[n * 9 + km], lo[km] * (1.0f / 64.0f));
  }
  int chead = nh * HC_;
  for (int h = 0; h < HC_; ++h) {
    float o = 0.0f;
    #pragma unroll
    for (int km = 0; km < 9; ++km) {
      long long vi = ((long long)(cbase + h) * 9 + km) * 2048 + n;
      o += lo[km] * (float)vv[vi];
    }
    out_mat[((long long)(b * 2048 + n)) * 256 + chead + h] = o;
  }
}

// ---------------------------------------------------------------------------
// Final loc layer (N=2, too small for WMMA tile): pred = l2 @ Wl3^T + bl3
// ---------------------------------------------------------------------------
__global__ void loc_final_kernel(const float* __restrict__ l2,
                                 const float* __restrict__ Wl3,
                                 const float* __restrict__ bl3,
                                 float* __restrict__ pred)
{
  int t = blockIdx.x * 256 + threadIdx.x;        // 16384*2
  int o = t & 1, r = t >> 1;
  float acc = bl3[o];
  const float* row = l2 + (long long)r * 256;
  const float* wr  = Wl3 + o * 256;
  for (int c = 0; c < 256; ++c) acc += row[c] * wr[c];
  pred[t] = acc;
}

// ---------------------------------------------------------------------------
// Losses + output zeroing
// ---------------------------------------------------------------------------
__global__ void zero_kernel(float* __restrict__ losses, float* __restrict__ attn_mean)
{
  int t = blockIdx.x * 256 + threadIdx.x;
  if (t < 2) losses[t] = 0.0f;
  if (t < 2048 * 9) attn_mean[t] = 0.0f;
}

__global__ void loss_cls_kernel(const float* __restrict__ pred,
                                const int* __restrict__ labels,
                                float* __restrict__ loss)
{
  __shared__ float red[256];
  int t = blockIdx.x * 256 + threadIdx.x;
  int r = t >> 8, j = t & 255;
  float p = pred[t];
  float tt = (labels[r] == j) ? 1.0f : 0.0f;
  float v = fmaxf(p, 0.0f) - p * tt + log1pf(expf(-fabsf(p)));
  red[threadIdx.x] = v;
  __syncthreads();
  for (int s = 128; s > 0; s >>= 1) {
    if (threadIdx.x < s) red[threadIdx.x] += red[threadIdx.x + s];
    __syncthreads();
  }
  if (threadIdx.x == 0) atomicAdd(loss, red[0] * (1.0f / 4194304.0f));
}

__global__ void loss_loc_kernel(const float* __restrict__ pred,
                                const float* __restrict__ coords,
                                float* __restrict__ loss)
{
  __shared__ float red[256];
  int t = blockIdx.x * 256 + threadIdx.x;
  float v = fabsf(pred[t] - coords[t]);
  red[threadIdx.x] = v;
  __syncthreads();
  for (int s = 128; s > 0; s >>= 1) {
    if (threadIdx.x < s) red[threadIdx.x] += red[threadIdx.x + s];
    __syncthreads();
  }
  if (threadIdx.x == 0) atomicAdd(loss, red[0] * (1.0f / 32768.0f));
}

// ---------------------------------------------------------------------------
// Host-side orchestration. Workspace layout (bytes) — ~313 MiB total.
// ---------------------------------------------------------------------------
extern "C" void kernel_launch(void* const* d_in, const int* in_sizes, int n_in,
                              void* d_out, int out_size, void* d_ws, size_t ws_size,
                              hipStream_t stream)
{
  const float* x    = (const float*)d_in[0];
  const int*   idx  = (const int*)d_in[1];
  const int*   labels = (const int*)d_in[2];
  const float* coords = (const float*)d_in[3];
  const float* W_ip = (const float*)d_in[4];
  const float* b_ip = (const float*)d_in[5];
  const float* Wq   = (const float*)d_in[6];
  const float* bq   = (const float*)d_in[7];
  const float* Wk   = (const float*)d_in[8];
  const float* bk   = (const float*)d_in[9];
  const float* Wv   = (const float*)d_in[10];
  const float* bv   = (const float*)d_in[11];
  const float* Wdw  = (const float*)d_in[12];
  const float* bdw  = (const float*)d_in[13];
  const float* Wpw  = (const float*)d_in[14];
  const float* Wc1  = (const float*)d_in[15];
  const float* bc1  = (const float*)d_in[16];
  const float* Wc2  = (const float*)d_in[17];
  const float* bc2  = (const float*)d_in[18];
  const float* Wc3  = (const float*)d_in[19];
  const float* bc3  = (const float*)d_in[20];
  const float* Wl1  = (const float*)d_in[21];
  const float* bl1  = (const float*)d_in[22];
  const float* Wl2  = (const float*)d_in[23];
  const float* bl2  = (const float*)d_in[24];
  const float* Wl3  = (const float*)d_in[25];
  const float* bl3  = (const float*)d_in[26];

  char* ws = (char*)d_ws;
  float*  ws_x1  = (float*)(ws + 0LL);
  float*  ws_qf  = (float*)(ws + 33554432LL);
  float*  ws_h   = (float*)(ws + 67108864LL);
  float*  ws_pos = (float*)(ws + 100663296LL);
  float*  ws_q   = (float*)(ws + 110100480LL);
  __bf16* ws_kk  = (__bf16*)(ws + 126877696LL);
  __bf16* ws_vv  = (__bf16*)(ws + 202375168LL);
  float*  ws_out = (float*)(ws + 277872640LL);
  float*  ws_h1  = (float*)(ws + 294649856LL);
  float*  ws_h2  = (float*)(ws + 311427072LL);

  float* out = (float*)d_out;
  float* out_xs   = out + 2;                 // 37,748,736
  float* out_cls  = out + 37748738LL;        //  4,194,304
  float* out_loc  = out + 41943042LL;        //     32,768
  float* out_attn = out + 41975810LL;        //     18,432

  dim3 blk(256);

  // 0) zero accumulated outputs (losses + attn mean)
  zero_kernel<<<dim3(72), blk, 0, stream>>>(out, out_attn);

  // 1) x1 = gelu(W_ip @ x + b_ip)   M=256 N=4096 K=256, batch 8
  gemm_bf16_wmma<<<dim3(32, 2, 8), blk, 0, stream>>>(
      W_ip, 0LL, 256, x, (long long)NC_ * HW_, HW_, 1,
      ws_x1, (long long)NC_ * HW_, HW_, b_ip, 1, 1, 0, 256, 4096, 256);

  // 2) qf = Wq @ x1 + bq
  gemm_bf16_wmma<<<dim3(32, 2, 8), blk, 0, stream>>>(
      Wq, 0LL, 256, ws_x1, (long long)NC_ * HW_, HW_, 1,
      ws_qf, (long long)NC_ * HW_, HW_, bq, 1, 0, 0, 256, 4096, 256);

  // 3) depthwise conv + gelu
  dwconv_kernel<<<dim3(32768), blk, 0, stream>>>(ws_qf, Wdw, bdw, ws_h);

  // 4) offsets -> pos
  offset_kernel<<<dim3(512), blk, 0, stream>>>(ws_h, Wpw, ws_pos);

  // 5) gather q
  qgather_kernel<<<dim3(16384), blk, 0, stream>>>(ws_qf, idx, ws_q);

  // 6) bilinear sample -> x_sampled (output region)
  sample_kernel<<<dim3(2304), blk, 0, stream>>>(ws_x1, ws_pos, idx, out_xs);

  // 7/8) K,V projections  M=256 N=18432 K=256, batch 8, bf16 output
  gemm_bf16_wmma<<<dim3(144, 2, 8), blk, 0, stream>>>(
      Wk, 0LL, 256, out_xs, (long long)NC_ * KM_ * NS_, KM_ * NS_, 1,
      ws_kk, (long long)NC_ * KM_ * NS_, KM_ * NS_, bk, 1, 0, 1, 256, 18432, 256);
  gemm_bf16_wmma<<<dim3(144, 2, 8), blk, 0, stream>>>(
      Wv, 0LL, 256, out_xs, (long long)NC_ * KM_ * NS_, KM_ * NS_, 1,
      ws_vv, (long long)NC_ * KM_ * NS_, KM_ * NS_, bv, 1, 0, 1, 256, 18432, 256);

  // 9) attention -> out_mat + attn mean
  attention_kernel<<<dim3(512), blk, 0, stream>>>(ws_q, ws_kk, ws_vv, ws_out, out_attn);

  // 10-12) cls MLP: relu, relu, linear   M=16384 N=256 K=256 (B = W^T)
  gemm_bf16_wmma<<<dim3(2, 128, 1), blk, 0, stream>>>(
      ws_out, 0LL, 256, Wc1, 0LL, 1, 256,
      ws_h1, 0LL, 256, bc1, 2, 2, 0, 16384, 256, 256);
  gemm_bf16_wmma<<<dim3(2, 128, 1), blk, 0, stream>>>(
      ws_h1, 0LL, 256, Wc2, 0LL, 1, 256,
      ws_h2, 0LL, 256, bc2, 2, 2, 0, 16384, 256, 256);
  gemm_bf16_wmma<<<dim3(2, 128, 1), blk, 0, stream>>>(
      ws_h2, 0LL, 256, Wc3, 0LL, 1, 256,
      out_cls, 0LL, 256, bc3, 2, 0, 0, 16384, 256, 256);

  // 13-15) loc MLP (reuse h1/h2), final N=2 layer in VALU
  gemm_bf16_wmma<<<dim3(2, 128, 1), blk, 0, stream>>>(
      ws_out, 0LL, 256, Wl1, 0LL, 1, 256,
      ws_h1, 0LL, 256, bl1, 2, 2, 0, 16384, 256, 256);
  gemm_bf16_wmma<<<dim3(2, 128, 1), blk, 0, stream>>>(
      ws_h1, 0LL, 256, Wl2, 0LL, 1, 256,
      ws_h2, 0LL, 256, bl2, 2, 2, 0, 16384, 256, 256);
  loc_final_kernel<<<dim3(128), blk, 0, stream>>>(ws_h2, Wl3, bl3, out_loc);

  // 16/17) losses
  loss_cls_kernel<<<dim3(16384), blk, 0, stream>>>(out_cls, labels, out + 0);
  loss_loc_kernel<<<dim3(128), blk, 0, stream>>>(out_loc, coords, out + 1);
}
